// MultiHeadAttention_36438502539497
// MI455X (gfx1250) — compile-verified, matrix operand order checked by reading, not run on hardware
//
#include <hip/hip_runtime.h>
#include <hip/hip_bf16.h>

// ---------------------------------------------------------------------------
// MHA forward for MI455X (gfx1250, wave32, WMMA).
// All GEMMs: v_wmma_f32_16x16x32_bf16 (bf16 A/B, f32 accumulate).
// Per-wave tile: 2 A-frags x 4 B-frags = 8 WMMA per k-step from 12 ds_load_b128.
// One-deep software pipeline: global loads for k+1 are issued while WMMAs for
// k run; register->LDS stores happen at the top of the next iteration.
// f32->bf16 conversion uses hardware packed cvt (via __builtin_convertvector)
// instead of manual bit-twiddled RNE.
//
// Fragment LDS layout per CDNA5 ISA 7.12.2 (16-bit, 16x32):
//   elem e of lane l -> k = (e%8) + (e/8)*16 + (l/16)*8, m(or n) = l%16.
//   C/D f32 16x16: vgpr r, lane l -> m = r + (l/16)*8, n = l%16.
// Tiles staged in fragment order: 32 lanes x 16 bf16 = 256 dwords per slot,
// so each lane fetches a fragment as one 32B contiguous chunk (2x ds_load_b128).
// ---------------------------------------------------------------------------

typedef __attribute__((ext_vector_type(16))) __bf16 v16bf;
typedef __attribute__((ext_vector_type(2)))  __bf16 v2bf;
typedef __attribute__((ext_vector_type(8)))  float  v8f;
typedef __attribute__((ext_vector_type(2)))  float  v2f;

#define SEQ   2048
#define DIM   1024
#define HEADS 16
#define HDIM  64
#define MROWS 8192            // B*S
#define BK    32

__device__ __forceinline__ unsigned short f2bf(float f) {
  __bf16 h = (__bf16)f;                       // RNE fptrunc -> v_cvt
  return __builtin_bit_cast(unsigned short, h);
}
__device__ __forceinline__ unsigned pack2(float a, float b) {
  v2f in = {a, b};
  v2bf out = __builtin_convertvector(in, v2bf);   // packed cvt
  return __builtin_bit_cast(unsigned, out);
}

// fragment-slot dword offset (slot = 256 dwords = 32 lanes x 16 bf16)
__device__ __forceinline__ int frag_dword(int slot, int row16, int k4) {
  int lane = row16 + (((k4 >> 3) & 1) << 4);
  int e0   = (k4 & 7) + ((k4 >> 4) << 3);
  return slot * 256 + lane * 8 + (e0 >> 1);
}

// 8 WMMAs: A slots {2*wm, 2*wm+1}, B slots {4*wn .. 4*wn+3}
__device__ __forceinline__ void wmma8(const unsigned* ldsA, const unsigned* ldsB,
                                      int wm, int wn, int lane, v8f acc[2][4]) {
  v16bf a0 = *(const v16bf*)&ldsA[(2 * wm + 0) * 256 + lane * 8];
  v16bf a1 = *(const v16bf*)&ldsA[(2 * wm + 1) * 256 + lane * 8];
#pragma unroll
  for (int j = 0; j < 4; ++j) {
    v16bf b = *(const v16bf*)&ldsB[(4 * wn + j) * 256 + lane * 8];
    acc[0][j] = __builtin_amdgcn_wmma_f32_16x16x32_bf16(
        false, a0, false, b, (short)0, acc[0][j], false, false);
    acc[1][j] = __builtin_amdgcn_wmma_f32_16x16x32_bf16(
        false, a1, false, b, (short)0, acc[1][j], false, false);
  }
}

// ---------------------------------------------------------------------------
// Kernel 1: fused QKV projections.  C[m,n] = X[m,:] . W[n,:] + b[n]
// Block tile 128x128 (8 waves = 4M x 2N), grid (64, 8, 3).
// Output bf16, head-split [B,H,S,HD].
// ---------------------------------------------------------------------------
__global__ __launch_bounds__(256) void qkv_proj_kernel(
    const float* __restrict__ q_in, const float* __restrict__ k_in,
    const float* __restrict__ v_in,
    const float* __restrict__ Wq, const float* __restrict__ bq,
    const float* __restrict__ Wk, const float* __restrict__ bk,
    const float* __restrict__ Wv, const float* __restrict__ bv,
    unsigned short* __restrict__ ws)
{
  const int which = blockIdx.z;
  const float* X    = which == 0 ? q_in : (which == 1 ? k_in : v_in);
  const float* W    = which == 0 ? Wq   : (which == 1 ? Wk   : Wv);
  const float* bias = which == 0 ? bq   : (which == 1 ? bk   : bv);
  unsigned short* dst = ws + (size_t)which * ((size_t)MROWS * DIM);

  __shared__ unsigned ldsA[2048];   // 8 slots (128 rows)
  __shared__ unsigned ldsB[2048];   // 8 slots (128 cols)

  const int t = threadIdx.x, wave = t >> 5, lane = t & 31;
  const int wm = wave & 3, wn = wave >> 2;
  const int m0 = blockIdx.x * 128;
  const int n0 = blockIdx.y * 128;

  v8f acc[2][4];
  v8f zero = {};
#pragma unroll
  for (int i = 0; i < 2; ++i)
#pragma unroll
    for (int j = 0; j < 4; ++j) acc[i][j] = zero;

  float4 aR[4], bR[4];
  auto ldA = [&](int kb) {
#pragma unroll
    for (int it = 0; it < 4; ++it) {
      int g = t + it * 256, m = g >> 3, k4 = (g & 7) << 2;
      aR[it] = *(const float4*)(X + (size_t)(m0 + m) * DIM + kb + k4);
    }
  };
  auto ldB = [&](int kb) {
#pragma unroll
    for (int it = 0; it < 4; ++it) {
      int g = t + it * 256, n = g >> 3, k4 = (g & 7) << 2;
      bR[it] = *(const float4*)(W + (size_t)(n0 + n) * DIM + kb + k4);
    }
  };
  auto stAB = [&]() {
#pragma unroll
    for (int it = 0; it < 4; ++it) {
      int g = t + it * 256, r = g >> 3, k4 = (g & 7) << 2;
      unsigned* pa = &ldsA[frag_dword(r >> 4, r & 15, k4)];
      pa[0] = pack2(aR[it].x, aR[it].y);
      pa[1] = pack2(aR[it].z, aR[it].w);
      unsigned* pb = &ldsB[frag_dword(r >> 4, r & 15, k4)];
      pb[0] = pack2(bR[it].x, bR[it].y);
      pb[1] = pack2(bR[it].z, bR[it].w);
    }
  };

  ldA(0); ldB(0);
  for (int kb = 0; kb < DIM; kb += BK) {
    stAB();
    __syncthreads();
    if (kb + BK < DIM) { ldA(kb + BK); ldB(kb + BK); }   // prefetch next tile
    wmma8(ldsA, ldsB, wm, wn, lane, acc);
    __syncthreads();
  }

#pragma unroll
  for (int j = 0; j < 4; ++j) {
    int n = n0 + wn * 64 + j * 16 + (lane & 15);
    float bias_v = bias[n];
    int h = n >> 6, hd = n & 63;
#pragma unroll
    for (int i = 0; i < 2; ++i)
#pragma unroll
      for (int r = 0; r < 8; ++r) {
        int m = m0 + wm * 32 + i * 16 + r + ((lane >> 4) << 3);
        int b = m >> 11, s = m & (SEQ - 1);
        dst[((size_t)(b * HEADS + h) * SEQ + s) * HDIM + hd] =
            f2bf(acc[i][j][r] + bias_v);
      }
  }
}

// ---------------------------------------------------------------------------
// Kernel 2: scores = Q . K^T * 0.125 per (b,h).  Block tile 128x128,
// grid (16, 16, 64).  Writes raw f32 scores into attn region of d_out.
// ---------------------------------------------------------------------------
__global__ __launch_bounds__(256) void scores_kernel(
    const unsigned short* __restrict__ ws, float* __restrict__ attn)
{
  const unsigned short* Qb = ws;
  const unsigned short* Kb = ws + (size_t)MROWS * DIM;

  __shared__ unsigned ldsA[2048];
  __shared__ unsigned ldsB[2048];

  const int t = threadIdx.x, wave = t >> 5, lane = t & 31;
  const int wm = wave & 3, wn = wave >> 2;
  const int m0 = blockIdx.x * 128;
  const int n0 = blockIdx.y * 128;
  const size_t bh = blockIdx.z;
  const unsigned short* Qh = Qb + bh * (size_t)SEQ * HDIM;
  const unsigned short* Kh = Kb + bh * (size_t)SEQ * HDIM;

  v8f acc[2][4];
  v8f zero = {};
#pragma unroll
  for (int i = 0; i < 2; ++i)
#pragma unroll
    for (int j = 0; j < 4; ++j) acc[i][j] = zero;

  uint2 aR[4], bR[4];
  auto ldA = [&](int kb) {
#pragma unroll
    for (int it = 0; it < 4; ++it) {
      int g = t + it * 256, m = g >> 3, k4 = (g & 7) << 2;
      aR[it] = *(const uint2*)(Qh + (size_t)(m0 + m) * HDIM + kb + k4);
    }
  };
  auto ldB = [&](int kb) {
#pragma unroll
    for (int it = 0; it < 4; ++it) {
      int g = t + it * 256, n = g >> 3, k4 = (g & 7) << 2;
      bR[it] = *(const uint2*)(Kh + (size_t)(n0 + n) * HDIM + kb + k4);
    }
  };
  auto stAB = [&]() {
#pragma unroll
    for (int it = 0; it < 4; ++it) {
      int g = t + it * 256, r = g >> 3, k4 = (g & 7) << 2;
      unsigned* pa = &ldsA[frag_dword(r >> 4, r & 15, k4)];
      pa[0] = aR[it].x; pa[1] = aR[it].y;
      unsigned* pb = &ldsB[frag_dword(r >> 4, r & 15, k4)];
      pb[0] = bR[it].x; pb[1] = bR[it].y;
    }
  };

  ldA(0); ldB(0);
  for (int kb = 0; kb < HDIM; kb += BK) {
    stAB();
    __syncthreads();
    if (kb + BK < HDIM) { ldA(kb + BK); ldB(kb + BK); }
    wmma8(ldsA, ldsB, wm, wn, lane, acc);
    __syncthreads();
  }

  float* out = attn + bh * (size_t)SEQ * SEQ;
#pragma unroll
  for (int j = 0; j < 4; ++j) {
    int n = n0 + wn * 64 + j * 16 + (lane & 15);
#pragma unroll
    for (int i = 0; i < 2; ++i)
#pragma unroll
      for (int r = 0; r < 8; ++r) {
        int m = m0 + wm * 32 + i * 16 + r + ((lane >> 4) << 3);
        out[(size_t)m * SEQ + n] = acc[i][j][r] * 0.125f;   // 1/sqrt(64)
      }
  }
}

// ---------------------------------------------------------------------------
// Kernel 3: in-place row softmax over attn.  One block (256 thr) per row.
// ---------------------------------------------------------------------------
__global__ __launch_bounds__(256) void softmax_kernel(float* __restrict__ attn)
{
  const size_t row = blockIdx.x;
  float* p = attn + row * (size_t)SEQ;
  const int t = threadIdx.x;

  float v[8];
  float mx = -3.402823466e38f;
#pragma unroll
  for (int j = 0; j < 8; ++j) { v[j] = p[t + j * 256]; mx = fmaxf(mx, v[j]); }

  __shared__ float red[256];
  red[t] = mx; __syncthreads();
  for (int s = 128; s > 0; s >>= 1) {
    if (t < s) red[t] = fmaxf(red[t], red[t + s]);
    __syncthreads();
  }
  mx = red[0]; __syncthreads();

  float sum = 0.f;
#pragma unroll
  for (int j = 0; j < 8; ++j) { v[j] = __expf(v[j] - mx); sum += v[j]; }
  red[t] = sum; __syncthreads();
  for (int s = 128; s > 0; s >>= 1) {
    if (t < s) red[t] += red[t + s];
    __syncthreads();
  }
  float inv = 1.0f / red[0];
#pragma unroll
  for (int j = 0; j < 8; ++j) p[t + j * 256] = v[j] * inv;
}

// ---------------------------------------------------------------------------
// Kernel 4: context = P . V per (b,h).  Block tile 256x64 (8 waves stacked
// in M, each 32 rows x full 64-col N).  grid (8, 1, 64).
// P: f32 probs from d_out (bf16 on staging).  V: bf16 ws.
// Output: bf16 ctx in [B, S, H*HD] layout.
// ---------------------------------------------------------------------------
__global__ __launch_bounds__(256) void context_kernel(
    const float* __restrict__ attn, const unsigned short* __restrict__ Vb,
    unsigned short* __restrict__ ctx)
{
  __shared__ unsigned ldsA[4096];   // 16 slots (256 rows)
  __shared__ unsigned ldsB[1024];   // 4 slots (64 cols)

  const int t = threadIdx.x, wave = t >> 5, lane = t & 31;
  const int m0 = blockIdx.x * 256;
  const size_t bh = blockIdx.z;
  const float* P = attn + bh * (size_t)SEQ * SEQ;
  const unsigned short* Vh = Vb + bh * (size_t)SEQ * HDIM;

  v8f acc[2][4];
  v8f zero = {};
#pragma unroll
  for (int i = 0; i < 2; ++i)
#pragma unroll
    for (int j = 0; j < 4; ++j) acc[i][j] = zero;

  float4 aR[8];
  uint2  bR[2];
  auto ldA = [&](int kb) {
#pragma unroll
    for (int it = 0; it < 8; ++it) {
      int g = t + it * 256, m = g >> 3, k4 = (g & 7) << 2;
      aR[it] = *(const float4*)(P + (size_t)(m0 + m) * SEQ + kb + k4);
    }
  };
  auto ldB = [&](int kb) {
#pragma unroll
    for (int it = 0; it < 2; ++it) {
      int g = t + it * 256, k = g >> 4, n4 = (g & 15) << 2;
      bR[it] = *(const uint2*)(Vh + (size_t)(kb + k) * HDIM + n4);
    }
  };
  auto stAB = [&]() {
#pragma unroll
    for (int it = 0; it < 8; ++it) {
      int g = t + it * 256, m = g >> 3, k4 = (g & 7) << 2;
      unsigned* pa = &ldsA[frag_dword(m >> 4, m & 15, k4)];
      pa[0] = pack2(aR[it].x, aR[it].y);
      pa[1] = pack2(aR[it].z, aR[it].w);
    }
    // V[k][n]: n contiguous in memory -> scatter b16 stores into frag order
#pragma unroll
    for (int it = 0; it < 2; ++it) {
      int g = t + it * 256, k = g >> 4, n4 = (g & 15) << 2;
      unsigned short ev[4] = {
        (unsigned short)(bR[it].x & 0xffff), (unsigned short)(bR[it].x >> 16),
        (unsigned short)(bR[it].y & 0xffff), (unsigned short)(bR[it].y >> 16)};
      int e = (k & 7) + ((k >> 4) << 3);
      int laneHalf = ((k >> 3) & 1) << 4;
      unsigned short* lb = (unsigned short*)ldsB;
#pragma unroll
      for (int i = 0; i < 4; ++i) {
        int n = n4 + i;
        lb[((n >> 4) * 256 + ((n & 15) + laneHalf) * 8) * 2 + e] = ev[i];
      }
    }
  };

  ldA(0); ldB(0);
  for (int kb = 0; kb < SEQ; kb += BK) {
    stAB();
    __syncthreads();
    if (kb + BK < SEQ) { ldA(kb + BK); ldB(kb + BK); }
    wmma8(ldsA, ldsB, /*wm=*/wave, /*wn=*/0, lane, acc);
    __syncthreads();
  }

  const int b = (int)(bh >> 4), h = (int)(bh & 15);
#pragma unroll
  for (int j = 0; j < 4; ++j) {
    int hd = j * 16 + (lane & 15);
#pragma unroll
    for (int i = 0; i < 2; ++i)
#pragma unroll
      for (int r = 0; r < 8; ++r) {
        int sq = m0 + wave * 32 + i * 16 + r + ((lane >> 4) << 3);
        ctx[((size_t)(b * SEQ + sq)) * DIM + h * HDIM + hd] = f2bf(acc[i][j][r]);
      }
  }
}

// ---------------------------------------------------------------------------
// Kernel 5: out = ctx . Wo^T + bo.  Block tile 128x128, grid (64, 8).
// ---------------------------------------------------------------------------
__global__ __launch_bounds__(256) void out_proj_kernel(
    const unsigned short* __restrict__ ctx, const float* __restrict__ Wo,
    const float* __restrict__ bo, float* __restrict__ out)
{
  __shared__ unsigned ldsA[2048];
  __shared__ unsigned ldsB[2048];

  const int t = threadIdx.x, wave = t >> 5, lane = t & 31;
  const int wm = wave & 3, wn = wave >> 2;
  const int m0 = blockIdx.x * 128;
  const int n0 = blockIdx.y * 128;

  v8f acc[2][4];
  v8f zero = {};
#pragma unroll
  for (int i = 0; i < 2; ++i)
#pragma unroll
    for (int j = 0; j < 4; ++j) acc[i][j] = zero;

  uint2  aR[4];
  float4 bR[4];
  auto ldA = [&](int kb) {
#pragma unroll
    for (int it = 0; it < 4; ++it) {
      int g = t + it * 256, m = g >> 3, k4 = (g & 7) << 2;
      aR[it] = *(const uint2*)(ctx + (size_t)(m0 + m) * DIM + kb + k4);
    }
  };
  auto ldB = [&](int kb) {
#pragma unroll
    for (int it = 0; it < 4; ++it) {
      int g = t + it * 256, n = g >> 3, k4 = (g & 7) << 2;
      bR[it] = *(const float4*)(Wo + (size_t)(n0 + n) * DIM + kb + k4);
    }
  };
  auto stAB = [&]() {
#pragma unroll
    for (int it = 0; it < 4; ++it) {
      int g = t + it * 256, r = g >> 3, k4 = (g & 7) << 2;
      unsigned* pa = &ldsA[frag_dword(r >> 4, r & 15, k4)];
      pa[0] = aR[it].x; pa[1] = aR[it].y;
      unsigned* pb = &ldsB[frag_dword(r >> 4, r & 15, k4)];
      pb[0] = pack2(bR[it].x, bR[it].y);
      pb[1] = pack2(bR[it].z, bR[it].w);
    }
  };

  ldA(0); ldB(0);
  for (int kb = 0; kb < DIM; kb += BK) {
    stAB();
    __syncthreads();
    if (kb + BK < DIM) { ldA(kb + BK); ldB(kb + BK); }
    wmma8(ldsA, ldsB, wm, wn, lane, acc);
    __syncthreads();
  }

#pragma unroll
  for (int j = 0; j < 4; ++j) {
    int n = n0 + wn * 64 + j * 16 + (lane & 15);
    float bias_v = bo[n];
#pragma unroll
    for (int i = 0; i < 2; ++i)
#pragma unroll
      for (int r = 0; r < 8; ++r) {
        int m = m0 + wm * 32 + i * 16 + r + ((lane >> 4) << 3);
        out[(size_t)m * DIM + n] = acc[i][j][r] + bias_v;
      }
  }
}

// ---------------------------------------------------------------------------
// Launch.  d_out = [ out (8192*1024 f32) | attn (64*2048*2048 f32) ].
// d_ws: Qb | Kb | Vb | ctx, bf16, 64 MB total.
// ---------------------------------------------------------------------------
extern "C" void kernel_launch(void* const* d_in, const int* in_sizes, int n_in,
                              void* d_out, int out_size, void* d_ws, size_t ws_size,
                              hipStream_t stream) {
  const float* q  = (const float*)d_in[0];
  const float* k  = (const float*)d_in[1];
  const float* v  = (const float*)d_in[2];
  const float* Wq = (const float*)d_in[3];
  const float* bq = (const float*)d_in[4];
  const float* Wk = (const float*)d_in[5];
  const float* bk = (const float*)d_in[6];
  const float* Wv = (const float*)d_in[7];
  const float* bv = (const float*)d_in[8];
  const float* Wo = (const float*)d_in[9];
  const float* bo = (const float*)d_in[10];

  float* out  = (float*)d_out;
  float* attn = out + (size_t)MROWS * DIM;

  unsigned short* ws  = (unsigned short*)d_ws;
  unsigned short* Vb  = ws + 2ull * (size_t)MROWS * DIM;
  unsigned short* ctx = ws + 3ull * (size_t)MROWS * DIM;

  qkv_proj_kernel<<<dim3(MROWS / 128, DIM / 128, 3), 256, 0, stream>>>(
      q, k, v, Wq, bq, Wk, bk, Wv, bv, ws);
  scores_kernel<<<dim3(SEQ / 128, SEQ / 128, 64), 256, 0, stream>>>(ws, attn);
  softmax_kernel<<<dim3(64 * SEQ, 1, 1), 256, 0, stream>>>(attn);
  context_kernel<<<dim3(SEQ / 256, 1, 64), 256, 0, stream>>>(attn, Vb, ctx);
  out_proj_kernel<<<dim3(MROWS / 128, DIM / 128, 1), 256, 0, stream>>>(ctx, Wo, bo, out);
}
